// AttentionBlock_17197049053389
// MI455X (gfx1250) — compile-verified, matrix operand order checked by reading, not run on hardware
//
#include <hip/hip_runtime.h>

// ---------------- problem constants ----------------
constexpr int B  = 4;
constexpr int C  = 512;
constexpr int T  = 2048;
constexpr int H  = 8;
constexpr int CH = 64;          // C / H
constexpr int G  = 32;
constexpr float EPS   = 1e-5f;
constexpr float S4    = 0.3535533905932738f;   // 64^-0.25
constexpr float LOG2E = 1.4426950408889634f;

// ---------------- WMMA / TDM types ----------------
typedef __attribute__((ext_vector_type(16))) __bf16 v16bf;
typedef __attribute__((ext_vector_type(8)))  float  v8f;
typedef __attribute__((ext_vector_type(4)))  unsigned int v4u;
typedef __attribute__((ext_vector_type(8)))  int v8i;
typedef __attribute__((ext_vector_type(4)))  int v4i;

__device__ __forceinline__ v8f wmma_bf16(v16bf a, v16bf b, v8f c) {
  // D = A(16x32 bf16) * B(32x16 bf16) + C(16x16 f32)
  return __builtin_amdgcn_wmma_f32_16x16x32_bf16(
      /*neg_a=*/false, a, /*neg_b=*/false, b,
      /*c_mod=*/(short)0, c, /*reuse_a=*/false, /*reuse_b=*/false);
}

// round-to-nearest-even f32 -> bf16 (as raw u16)
__device__ __forceinline__ unsigned short f2bf(float f) {
  unsigned int u = __float_as_uint(f);
  u += 0x7FFFu + ((u >> 16) & 1u);
  return (unsigned short)(u >> 16);
}

// Load one 16x32 A/B WMMA operand.
// Memory layout: row-major [16 rows][K contiguous], `stride` elems between rows.
// Per ISA 16-bit A layout: lane L (r=L&15, hi=L>>4):
//   VGPR0..3 hold K = hi*8 + 0..7   -> one b128 at (r*stride + hi*8)
//   VGPR4..7 hold K = hi*8 + 16..23 -> one b128 at (r*stride + hi*8 + 16)
__device__ __forceinline__ v16bf load_tile(const unsigned short* base, int stride, int lane) {
  const int r = lane & 15, hi = lane >> 4;
  const unsigned short* p = base + (size_t)r * stride + hi * 8;
  union { v16bf v; uint4 u[2]; } x;
  x.u[0] = *(const uint4*)(p);
  x.u[1] = *(const uint4*)(p + 16);
  return x.v;
}

// Tensor Data Mover: 2D tile (16-bit elements) global -> LDS.
// D# layout per CDNA5 ISA. Groups 2/3 zero (<=2D tile).
// This toolchain exposes the 6-arg builtin:
//   (uint32x4 g0, int32x8 g1, int32x4 g2, int32x4 g3, int32x8 extra, i32 cpol)
__device__ __forceinline__ void tdm_load_2d(unsigned int lds_addr, const void* gptr,
                                            unsigned int tensor_d0, unsigned int tensor_d1,
                                            unsigned int stride0,   // dim0 stride, elements
                                            unsigned int tile_d0, unsigned int tile_d1) {
  unsigned long long ga = (unsigned long long)(uintptr_t)gptr;
  v4u g0;
  g0[0] = 1u;                                             // count=1, user descriptor
  g0[1] = lds_addr;                                       // LDS byte address
  g0[2] = (unsigned int)(ga & 0xFFFFFFFFu);               // global_addr[31:0]
  g0[3] = ((unsigned int)((ga >> 32) & 0x1FFFFFFull))     // global_addr[56:32]
          | 0x80000000u;                                  // type=2 ("image")
  v8i g1;
  g1[0] = (int)(1u << 16);                                // data_size = 1 -> 2 bytes
  g1[1] = (int)((tensor_d0 & 0xFFFFu) << 16);             // tensor_dim0[15:0] @ bits63:48
  g1[2] = (int)((tensor_d0 >> 16) | ((tensor_d1 & 0xFFFFu) << 16));
  g1[3] = (int)((tensor_d1 >> 16) | (tile_d0 << 16));     // tile_dim0 @ bits127:112
  g1[4] = (int)tile_d1;                                   // tile_dim1; tile_dim2=0
  g1[5] = (int)stride0;                                   // tensor_dim0_stride[31:0]
  g1[6] = 0;                                              // stride0[47:32] | stride1[15:0]
  g1[7] = 0;                                              // stride1[47:16]
  v4i g2 = {0, 0, 0, 0};
  v4i g3 = {0, 0, 0, 0};
  v8i g4 = {0, 0, 0, 0, 0, 0, 0, 0};
  __builtin_amdgcn_tensor_load_to_lds(g0, g1, g2, g3, g4, 0);
}

// reductions across the 16 lanes of a half-wave (wave32)
__device__ __forceinline__ float hmax16(float v) {
  v = fmaxf(v, __shfl_xor(v, 1, 32));
  v = fmaxf(v, __shfl_xor(v, 2, 32));
  v = fmaxf(v, __shfl_xor(v, 4, 32));
  v = fmaxf(v, __shfl_xor(v, 8, 32));
  return v;
}
__device__ __forceinline__ float hsum16(float v) {
  v += __shfl_xor(v, 1, 32);
  v += __shfl_xor(v, 2, 32);
  v += __shfl_xor(v, 4, 32);
  v += __shfl_xor(v, 8, 32);
  return v;
}

// ---------------- kernel 1: weight f32 -> bf16 ----------------
__global__ void convert_w_kernel(const float* __restrict__ qkv_w,
                                 const float* __restrict__ proj_w,
                                 unsigned short* __restrict__ wq,
                                 unsigned short* __restrict__ wp) {
  int i = blockIdx.x * 256 + threadIdx.x;
  if (i < 3 * C * C) wq[i] = f2bf(qkv_w[i]);
  if (i < C * C)     wp[i] = f2bf(proj_w[i]);
}

// ---------------- kernel 2: GroupNorm -> hT[b][t][c] bf16 ----------------
__global__ __launch_bounds__(256) void groupnorm_kernel(
    const float* __restrict__ x, const float* __restrict__ gscale,
    const float* __restrict__ gbias, unsigned short* __restrict__ hT) {
  const int b = blockIdx.x >> 5;          // / G
  const int g = blockIdx.x & 31;          // % G
  constexpr int CG = C / G;               // 16
  constexpr int N  = CG * T;              // 32768
  const float* xg = x + ((size_t)b * C + (size_t)g * CG) * T;

  float s = 0.f, s2 = 0.f;
  for (int i = threadIdx.x; i < N; i += 256) { float v = xg[i]; s += v; s2 += v * v; }
  const int lane = threadIdx.x & 31, wv = threadIdx.x >> 5;
#pragma unroll
  for (int m = 1; m < 32; m <<= 1) { s += __shfl_xor(s, m, 32); s2 += __shfl_xor(s2, m, 32); }
  __shared__ float rb[2][8];
  if (lane == 0) { rb[0][wv] = s; rb[1][wv] = s2; }
  __syncthreads();
  float ts = 0.f, ts2 = 0.f;
#pragma unroll
  for (int w = 0; w < 8; ++w) { ts += rb[0][w]; ts2 += rb[1][w]; }
  const float mean = ts / (float)N;
  const float var  = ts2 / (float)N - mean * mean;
  const float rinv = rsqrtf(var + EPS);

  for (int i = threadIdx.x; i < N; i += 256) {
    int cl = i >> 11;            // / T
    int t  = i & (T - 1);        // % T
    int c  = g * CG + cl;
    float y = (xg[i] - mean) * rinv * gscale[c] + gbias[c];
    hT[((size_t)b * T + t) * C + c] = f2bf(y);
  }
}

// ---------------- kernel 3: QKV GEMM (WMMA) ----------------
// qkv[b,o,t] = sum_c qkv_w[o,c] * h[b,c,t] + qkv_b[o]
// A = wq [o][c] row-major, B = hT [t][c] ("N-major, K contiguous")
__global__ __launch_bounds__(128) void qkv_gemm_kernel(
    const unsigned short* __restrict__ wq, const unsigned short* __restrict__ hT,
    const float* __restrict__ qkvb,
    unsigned short* __restrict__ qT, unsigned short* __restrict__ kT,
    unsigned short* __restrict__ vv) {
  const int b    = blockIdx.z;
  const int n0   = blockIdx.x * 64;                            // t tile
  const int mw   = blockIdx.y * 64 + (threadIdx.x >> 5) * 16;  // o rows for this wave
  const int lane = threadIdx.x & 31;
  const unsigned short* hTb = hT + (size_t)b * T * C;

  v8f acc[4] = {{0,0,0,0,0,0,0,0},{0,0,0,0,0,0,0,0},{0,0,0,0,0,0,0,0},{0,0,0,0,0,0,0,0}};
  for (int k0 = 0; k0 < C; k0 += 32) {
    v16bf a = load_tile(wq + (size_t)mw * C + k0, C, lane);
#pragma unroll
    for (int n = 0; n < 4; ++n) {
      v16bf bt = load_tile(hTb + (size_t)(n0 + n * 16) * C + k0, C, lane);
      acc[n] = wmma_bf16(a, bt, acc[n]);
    }
  }
  const int col = lane & 15, hi = lane >> 4;
#pragma unroll
  for (int n = 0; n < 4; ++n) {
    const int t = n0 + n * 16 + col;
#pragma unroll
    for (int r = 0; r < 8; ++r) {
      const int o = mw + r + 8 * hi;
      const float val = acc[n][r] + qkvb[o];
      if (o < C) {                     // q -> [bh][t][64], pre-scaled
        int hh = o >> 6, c = o & 63;
        qT[(((size_t)b * H + hh) * T + t) * CH + c] = f2bf(val * S4);
      } else if (o < 2 * C) {          // k -> [bh][t][64], pre-scaled
        int o2 = o - C; int hh = o2 >> 6, c = o2 & 63;
        kT[(((size_t)b * H + hh) * T + t) * CH + c] = f2bf(val * S4);
      } else {                         // v -> [bh][64][T]
        int o2 = o - 2 * C; int hh = o2 >> 6, c = o2 & 63;
        vv[(((size_t)b * H + hh) * CH + c) * T + t] = f2bf(val);
      }
    }
  }
}

// ---------------- kernel 4: flash attention (WMMA + TDM double buffering) ----
// per (bh, 64 query rows); wave owns 16 rows; s-tiles of 64 with online softmax.
// K/V s-tiles are DMA'd into LDS once per block by wave 0 via the Tensor Data
// Mover, double buffered, with s_wait_tensorcnt overlap.
__global__ __launch_bounds__(128) void attn_kernel(
    const unsigned short* __restrict__ qT, const unsigned short* __restrict__ kT,
    const unsigned short* __restrict__ vv, unsigned short* __restrict__ aT) {
  const int bh = blockIdx.y;
  const int b  = bh >> 3;
  const int h  = bh & 7;
  const int wave = threadIdx.x >> 5, lane = threadIdx.x & 31;
  const int tq = blockIdx.x * 64 + wave * 16;
  const int col = lane & 15, hi = lane >> 4;

  __shared__ __align__(16) unsigned short Kbuf[2][64][64];   // [t-rows][c], stride 64
  __shared__ __align__(16) unsigned short Vbuf[2][64][64];   // [c-rows][s], stride 64
  __shared__ __align__(16) unsigned short Pbuf[4][16][64];   // per-wave P tile

  const unsigned short* qB = qT + ((size_t)bh * T + tq) * CH;
  const unsigned short* kB = kT + (size_t)bh * T * CH;
  const unsigned short* vB = vv + (size_t)bh * CH * T;

  const v16bf aq0 = load_tile(qB, CH, lane);        // K = c 0..31
  const v16bf aq1 = load_tile(qB + 32, CH, lane);   // K = c 32..63

  v8f oacc[4] = {{0,0,0,0,0,0,0,0},{0,0,0,0,0,0,0,0},{0,0,0,0,0,0,0,0},{0,0,0,0,0,0,0,0}};
  float mi[8], li[8];
#pragma unroll
  for (int r = 0; r < 8; ++r) { mi[r] = -1e30f; li[r] = 0.f; }

  constexpr int NIT = T / 64;   // 32 s-tiles

  // initial K/V tile DMA into buffer 0 (wave 0 only; TDM ignores EXEC)
  if (wave == 0) {
    // K tile: 64 full rows of kT -> contiguous 4096 elements (1D)
    tdm_load_2d((unsigned int)(uintptr_t)&Kbuf[0][0][0], kB, 4096u, 1u, 4096u, 4096u, 0u);
    // V tile: 64 rows (c) x 64 cols (s), row stride T
    tdm_load_2d((unsigned int)(uintptr_t)&Vbuf[0][0][0], vB, (unsigned)T, 64u, (unsigned)T, 64u, 64u);
  }

  for (int it = 0; it < NIT; ++it) {
    const int bsel = it & 1;
    const int s0 = it * 64;
    if (wave == 0) {
      if (it + 1 < NIT) {
        // prefetch next tile into the other buffer, then require only the
        // current tile to be complete (prefetch may remain in flight)
        tdm_load_2d((unsigned int)(uintptr_t)&Kbuf[bsel ^ 1][0][0],
                    kB + (size_t)(s0 + 64) * CH, 4096u, 1u, 4096u, 4096u, 0u);
        tdm_load_2d((unsigned int)(uintptr_t)&Vbuf[bsel ^ 1][0][0],
                    vB + (s0 + 64), (unsigned)T, 64u, (unsigned)T, 64u, 64u);
        __builtin_amdgcn_s_wait_tensorcnt(2);
      } else {
        __builtin_amdgcn_s_wait_tensorcnt(0);
      }
    }
    __syncthreads();   // current buffer visible to all waves

    // S = q^T k  (16 t-rows x 64 s-cols), K from LDS
    v8f sacc[4];
#pragma unroll
    for (int n = 0; n < 4; ++n) {
      const unsigned short* kb = &Kbuf[bsel][n * 16][0];
      v16bf bk0 = load_tile(kb, 64, lane);
      v16bf bk1 = load_tile(kb + 32, 64, lane);
      v8f z = {0,0,0,0,0,0,0,0};
      z = wmma_bf16(aq0, bk0, z);
      z = wmma_bf16(aq1, bk1, z);
      sacc[n] = z;
    }
    // online softmax (row = r + 8*hi, spread across the 16 lanes of a half)
#pragma unroll
    for (int r = 0; r < 8; ++r) {
      float mnew = mi[r];
#pragma unroll
      for (int n = 0; n < 4; ++n) mnew = fmaxf(mnew, sacc[n][r]);
      mnew = hmax16(mnew);
      const float al = exp2f((mi[r] - mnew) * LOG2E);
      float ps = 0.f;
#pragma unroll
      for (int n = 0; n < 4; ++n) {
        float pv = exp2f((sacc[n][r] - mnew) * LOG2E);
        sacc[n][r] = pv;
        ps += pv;
      }
      ps = hsum16(ps);
      li[r] = li[r] * al + ps;
      mi[r] = mnew;
#pragma unroll
      for (int ct = 0; ct < 4; ++ct) oacc[ct][r] *= al;
    }
    // C-layout P -> LDS -> A-layout P (per-wave tile; same-wave DS ordering)
    unsigned short* Pw = &Pbuf[wave][0][0];
#pragma unroll
    for (int n = 0; n < 4; ++n)
#pragma unroll
      for (int r = 0; r < 8; ++r)
        Pw[(r + 8 * hi) * 64 + n * 16 + col] = f2bf(sacc[n][r]);
    v16bf p0 = load_tile(Pw, 64, lane);
    v16bf p1 = load_tile(Pw + 32, 64, lane);
    // O += P * V^T   (B = v [c][s] from LDS, K = s contiguous)
#pragma unroll
    for (int ct = 0; ct < 4; ++ct) {
      const unsigned short* vb = &Vbuf[bsel][ct * 16][0];
      v16bf bv0 = load_tile(vb, 64, lane);
      v16bf bv1 = load_tile(vb + 32, 64, lane);
      oacc[ct] = wmma_bf16(p0, bv0, oacc[ct]);
      oacc[ct] = wmma_bf16(p1, bv1, oacc[ct]);
    }
    __syncthreads();   // all waves done reading this buffer before it is re-filled
  }
  // normalize and write aT[b][t][c_global] bf16
#pragma unroll
  for (int r = 0; r < 8; ++r) {
    const float inv = 1.0f / li[r];
    const int t = tq + r + 8 * hi;
#pragma unroll
    for (int ct = 0; ct < 4; ++ct) {
      const int cg = h * CH + ct * 16 + col;
      aT[((size_t)b * T + t) * C + cg] = f2bf(oacc[ct][r] * inv);
    }
  }
}

// ---------------- kernel 5: proj GEMM + bias + residual ----------------
// out[b,o,t] = x[b,o,t] + proj_b[o] + sum_c aT[b][t][c] * proj_w[o][c]
__global__ __launch_bounds__(128) void proj_kernel(
    const unsigned short* __restrict__ aT, const unsigned short* __restrict__ wp,
    const float* __restrict__ projb, const float* __restrict__ x,
    float* __restrict__ out) {
  const int b    = blockIdx.z;
  const int n0   = blockIdx.x * 64;                            // o tile
  const int mw   = blockIdx.y * 64 + (threadIdx.x >> 5) * 16;  // t rows for this wave
  const int lane = threadIdx.x & 31;
  const unsigned short* aTb = aT + (size_t)b * T * C;

  v8f acc[4] = {{0,0,0,0,0,0,0,0},{0,0,0,0,0,0,0,0},{0,0,0,0,0,0,0,0},{0,0,0,0,0,0,0,0}};
  for (int k0 = 0; k0 < C; k0 += 32) {
    v16bf a = load_tile(aTb + (size_t)mw * C + k0, C, lane);
#pragma unroll
    for (int n = 0; n < 4; ++n) {
      v16bf bt = load_tile(wp + (size_t)(n0 + n * 16) * C + k0, C, lane);
      acc[n] = wmma_bf16(a, bt, acc[n]);
    }
  }
  const int col = lane & 15, hi = lane >> 4;
#pragma unroll
  for (int n = 0; n < 4; ++n) {
    const int o = n0 + n * 16 + col;
    const float pb = projb[o];
#pragma unroll
    for (int r = 0; r < 8; ++r) {
      const int t = mw + r + 8 * hi;
      const size_t idx = ((size_t)b * C + o) * T + t;
      out[idx] = x[idx] + acc[n][r] + pb;
    }
  }
}

// ---------------- launcher ----------------
extern "C" void kernel_launch(void* const* d_in, const int* in_sizes, int n_in,
                              void* d_out, int out_size, void* d_ws, size_t ws_size,
                              hipStream_t stream) {
  const float* x          = (const float*)d_in[0];
  const float* norm_scale = (const float*)d_in[1];
  const float* norm_bias  = (const float*)d_in[2];
  const float* qkv_w      = (const float*)d_in[3];
  const float* qkv_b      = (const float*)d_in[4];
  const float* proj_w     = (const float*)d_in[5];
  const float* proj_b     = (const float*)d_in[6];
  float* out = (float*)d_out;

  unsigned short* ws = (unsigned short*)d_ws;
  size_t off = 0;
  unsigned short* wq = ws + off; off += (size_t)3 * C * C;   // 1536x512 bf16
  unsigned short* wp = ws + off; off += (size_t)C * C;       // 512x512 bf16
  unsigned short* hT = ws + off; off += (size_t)B * T * C;   // [b][t][c]
  unsigned short* qT = ws + off; off += (size_t)B * T * C;   // [bh][t][64]
  unsigned short* kT = ws + off; off += (size_t)B * T * C;   // [bh][t][64]
  unsigned short* vv = ws + off; off += (size_t)B * T * C;   // [bh][64][T]
  unsigned short* aT = hT;                                   // reuse hT after qkv gemm

  convert_w_kernel<<<(3 * C * C) / 256, 256, 0, stream>>>(qkv_w, proj_w, wq, wp);
  groupnorm_kernel<<<B * G, 256, 0, stream>>>(x, norm_scale, norm_bias, hT);
  qkv_gemm_kernel<<<dim3(T / 64, (3 * C) / 64, B), 128, 0, stream>>>(wq, hT, qkv_b, qT, kT, vv);
  attn_kernel<<<dim3(T / 64, B * H), 128, 0, stream>>>(qT, kT, vv, aT);
  proj_kernel<<<dim3(C / 64, T / 64, B), 128, 0, stream>>>(aT, wp, proj_b, x, out);
}
